// GATv2_88261577932900
// MI455X (gfx1250) — matrix operand order repaired and checked by reading, not compiled, not run on hardware
//
#include <hip/hip_runtime.h>
#include <hip/hip_bf16.h>

// ---------------------------------------------------------------------------
// GATv2 (2 layers) for MI455X / gfx1250.
//   N=100000 nodes, E=1.6M edges, D=512, H1=8,F1=8 (hidden 64), C=40.
// Projections: v_wmma_f32_16x16x32_bf16. A staged to LDS as bf16 once per
// block (shared by all 8 waves); B pre-packed to WMMA lane layout in a
// one-shot kernel. Edge softmax/aggregation uses global atomics.
// ---------------------------------------------------------------------------

typedef __attribute__((ext_vector_type(16))) __bf16          v16bf;
typedef __attribute__((ext_vector_type(8)))  float           v8f;
typedef __attribute__((ext_vector_type(8)))  unsigned short  u16x8;

union ABPack { v16bf v; unsigned short u[16]; };

__device__ __forceinline__ unsigned short f2bf(float f) {
    unsigned u = __float_as_uint(f);
    // round-to-nearest-even bf16
    return (unsigned short)((u + 0x7FFFu + ((u >> 16) & 1u)) >> 16);
}

// monotone float <-> uint encoding for atomicMax on floats
__device__ __forceinline__ unsigned fenc(float f) {
    unsigned u = __float_as_uint(f);
    return (u & 0x80000000u) ? ~u : (u | 0x80000000u);
}
__device__ __forceinline__ float fdec(unsigned u) {
    return (u & 0x80000000u) ? __uint_as_float(u & 0x7FFFFFFFu)
                             : __uint_as_float(~u);
}

// ---------------------------------------------------------------------------
// One-shot weight pack: [Wa | Wb] (K x N1 / K x N2 fp32, fused columns) into
// WMMA B lane layout, bf16:
//   Bpk[((t*ksteps + ks)*32 + lane)*16 + i] = bf16(W[(ks*32 + khalf*16 + i), col])
//   col = t*16 + (lane&15), khalf = lane>>4
// Grid: (ctiles, ksteps) x 32 threads.
// ---------------------------------------------------------------------------
__global__ void gat_pack_w(const float* __restrict__ Wa, const float* __restrict__ Wb,
                           unsigned short* __restrict__ Bpk, int K, int N1, int N2)
{
    const int t    = blockIdx.x;
    const int ks   = blockIdx.y;
    const int lane = threadIdx.x;           // 0..31
    const int c     = (t << 4) + (lane & 15);
    const int khalf = lane >> 4;
    const float* Wp; int cc, ld;
    if (c < N1) { Wp = Wa; cc = c;      ld = N1; }
    else        { Wp = Wb; cc = c - N1; ld = N2; }
    const int ksteps = K >> 5;
    unsigned short* out = Bpk + (((long long)t * ksteps + ks) * 32 + lane) * 16;
    const int kbase = ks * 32 + khalf * 16;
    #pragma unroll
    for (int i = 0; i < 16; ++i)
        out[i] = f2bf(Wp[(long long)(kbase + i) * ld + cc]);
}

// ---------------------------------------------------------------------------
// Fused projection GEMM: Y[M, N1+N2] = X[M,K] @ Bpk + [ba | bb]
// Requires M % 16 == 0, K % 32 == 0. 256 threads = 8 waves; block owns a
// 16-row slab; each wave computes one 16x16 output tile.
// Dynamic LDS: 16 * (K + 8) bf16.
// ---------------------------------------------------------------------------
__global__ __launch_bounds__(256) void gat_wmma_proj(
    const float* __restrict__ X, const unsigned short* __restrict__ Bpk,
    const float* __restrict__ ba, const float* __restrict__ bb,
    float* __restrict__ Y, int K, int N1, int N2)
{
    extern __shared__ unsigned short Ash[];     // [16][K+8] bf16
    const int NO     = N1 + N2;
    const int ctiles = (NO + 15) >> 4;
    const int ksteps = K >> 5;
    const int stride = K + 8;                   // keeps 16B alignment (pad = 16B)
    const long long rowBase = (long long)blockIdx.x * 16;

    // ---- stage + convert the 16 x K A-slab into LDS (all 256 threads) ----
    {
        const int rr = threadIdx.x >> 4;                  // 0..15
        const int k0 = (threadIdx.x & 15) * (K >> 4);     // contiguous chunk
        const float* xp = X + (rowBase + rr) * K + k0;
        unsigned short* ap = Ash + rr * stride + k0;
        for (int i = 0; i < (K >> 4); ++i) ap[i] = f2bf(xp[i]);
    }
    __syncthreads();

    const int wave = threadIdx.x >> 5;
    const int lane = threadIdx.x & 31;
    if (wave >= ctiles) return;                 // wave-uniform exit

    const int r     = lane & 15;
    const int khalf = lane >> 4;

    const unsigned short* arow = Ash + r * stride;
    const unsigned short* bpk  = Bpk + (long long)wave * ksteps * 512;  // *32*16

    v8f acc = {};
    for (int ks = 0; ks < ksteps; ++ks) {
        const int kb = ks * 32;
        ABPack A, B;
        // A fragment: two 16B chunks from LDS (ds_load_b128 x2)
        *(u16x8*)&A.u[0] = *(const u16x8*)&arow[kb + khalf * 16];
        *(u16x8*)&A.u[8] = *(const u16x8*)&arow[kb + khalf * 16 + 8];
        // B fragment: 32B contiguous per lane (global_load_b128 x2, coalesced)
        const unsigned short* bp = bpk + (ks * 32 + lane) * 16;
        *(u16x8*)&B.u[0] = *(const u16x8*)&bp[0];
        *(u16x8*)&B.u[8] = *(const u16x8*)&bp[8];

        acc = __builtin_amdgcn_wmma_f32_16x16x32_bf16(
            false, A.v, false, B.v, (short)0, acc, false, false);
    }

    // ---- store D (16x16 f32): VGPR i -> M = i + 8*khalf, N = lane&15 ----
    const int c = (wave << 4) + r;
    const float bias = (c < N1) ? ba[c] : bb[c - N1];
    const long long rb = rowBase + khalf * 8;
    #pragma unroll
    for (int i = 0; i < 8; ++i)
        Y[(rb + i) * NO + c] = acc[i] + bias;
}

// ---------------------------------------------------------------------------
__global__ void gat_zero(float* __restrict__ p, long long n) {
    long long i = (long long)blockIdx.x * blockDim.x + threadIdx.x;
    if (i < n) p[i] = 0.0f;
}

// ---- layer 1 edge kernels (H=8, F=8, fs = Y[:, :64], fd = Y[:, 64:128]) ----
__global__ void gat_edge_score1(const float* __restrict__ Y,
                                const int* __restrict__ src, const int* __restrict__ dst,
                                const float* __restrict__ a1,
                                float* __restrict__ score, unsigned* __restrict__ menc, int E)
{
    int e = blockIdx.x * blockDim.x + threadIdx.x;
    if (e >= E) return;
    const int s = src[e], d = dst[e];
    const float* fs = Y + (long long)s * 128;
    const float* fd = Y + (long long)d * 128 + 64;
    #pragma unroll
    for (int h = 0; h < 8; ++h) {
        float sc = 0.f;
        #pragma unroll
        for (int f = 0; f < 8; ++f) {
            float v = fs[h * 8 + f] + fd[h * 8 + f];
            v = (v > 0.f) ? v : 0.2f * v;           // leaky relu
            sc += a1[h * 8 + f] * v;
        }
        score[(long long)e * 8 + h] = sc;
        atomicMax(&menc[(long long)d * 8 + h], fenc(sc));
    }
}

__global__ void gat_edge_acc1(const float* __restrict__ Y,
                              const int* __restrict__ src, const int* __restrict__ dst,
                              const float* __restrict__ score, const unsigned* __restrict__ menc,
                              float* __restrict__ dsum, float* __restrict__ acc, int E)
{
    int e = blockIdx.x * blockDim.x + threadIdx.x;
    if (e >= E) return;
    const int s = src[e], d = dst[e];
    const float* fs = Y + (long long)s * 128;
    #pragma unroll
    for (int h = 0; h < 8; ++h) {
        float m = fdec(menc[(long long)d * 8 + h]);
        float w = __expf(score[(long long)e * 8 + h] - m);
        atomicAdd(&dsum[(long long)d * 8 + h], w);
        #pragma unroll
        for (int f = 0; f < 8; ++f)
            atomicAdd(&acc[(long long)d * 64 + h * 8 + f], w * fs[h * 8 + f]);
    }
}

__global__ void gat_final1(float* __restrict__ acc, const float* __restrict__ dsum, long long n)
{
    long long i = (long long)blockIdx.x * blockDim.x + threadIdx.x;
    if (i >= n) return;
    long long node = i >> 6;          // /64
    int h = (int)((i & 63) >> 3);     // /8
    float v = acc[i] / dsum[node * 8 + h];
    acc[i] = (v > 0.f) ? v : (__expf(v) - 1.f);   // ELU (in place -> hidden h)
}

// ---- layer 2 edge kernels (H=1, F=40, fs2 = Y[:, :40], fd2 = Y[:, 40:80]) --
__global__ void gat_edge_score2(const float* __restrict__ Y,
                                const int* __restrict__ src, const int* __restrict__ dst,
                                const float* __restrict__ a2,
                                float* __restrict__ score, unsigned* __restrict__ menc, int E)
{
    int e = blockIdx.x * blockDim.x + threadIdx.x;
    if (e >= E) return;
    const int s = src[e], d = dst[e];
    const float* fs = Y + (long long)s * 80;
    const float* fd = Y + (long long)d * 80 + 40;
    float sc = 0.f;
    #pragma unroll
    for (int c = 0; c < 40; ++c) {
        float v = fs[c] + fd[c];
        v = (v > 0.f) ? v : 0.2f * v;
        sc += a2[c] * v;
    }
    score[e] = sc;
    atomicMax(&menc[d], fenc(sc));
}

__global__ void gat_edge_acc2(const float* __restrict__ Y,
                              const int* __restrict__ src, const int* __restrict__ dst,
                              const float* __restrict__ score, const unsigned* __restrict__ menc,
                              float* __restrict__ dsum, float* __restrict__ acc, int E)
{
    int e = blockIdx.x * blockDim.x + threadIdx.x;
    if (e >= E) return;
    const int s = src[e], d = dst[e];
    const float* fs = Y + (long long)s * 80;
    float w = __expf(score[e] - fdec(menc[d]));
    atomicAdd(&dsum[d], w);
    #pragma unroll
    for (int c = 0; c < 40; ++c)
        atomicAdd(&acc[(long long)d * 40 + c], w * fs[c]);
}

__global__ void gat_final2(const float* __restrict__ acc, const float* __restrict__ dsum,
                           float* __restrict__ out, long long n)
{
    long long i = (long long)blockIdx.x * blockDim.x + threadIdx.x;
    if (i >= n) return;
    out[i] = acc[i] / dsum[i / 40];
}

// ---------------------------------------------------------------------------
extern "C" void kernel_launch(void* const* d_in, const int* in_sizes, int n_in,
                              void* d_out, int out_size, void* d_ws, size_t ws_size,
                              hipStream_t stream)
{
    const float* x   = (const float*)d_in[0];
    const int*   src = (const int*)  d_in[1];
    const int*   dst = (const int*)  d_in[2];
    const float* W1s = (const float*)d_in[3];
    const float* b1s = (const float*)d_in[4];
    const float* W1d = (const float*)d_in[5];
    const float* b1d = (const float*)d_in[6];
    const float* a1  = (const float*)d_in[7];
    const float* W2s = (const float*)d_in[8];
    const float* b2s = (const float*)d_in[9];
    const float* W2d = (const float*)d_in[10];
    const float* b2d = (const float*)d_in[11];
    const float* a2  = (const float*)d_in[12];
    float* out = (float*)d_out;

    const int D = 512;
    const long long N = in_sizes[0] / D;   // 100000
    const int E = in_sizes[1];             // 1600000

    // workspace layout (floats); regions reused across the two layers
    float*    Y     = (float*)d_ws;               // N*128 (layer1) / N*80 (layer2)
    float*    score = Y + N * 128;                // E*8 (layer1) / E (layer2)
    unsigned* menc  = (unsigned*)(score + (long long)E * 8);  // N*8 / N
    float*    dsum  = (float*)menc + N * 8;       // N*8 / N
    float*    acc   = dsum + N * 8;               // N*64 (-> hidden h) / N*40
    unsigned short* wpk1 = (unsigned short*)(acc + N * 64);   // 8*16*32*16 bf16
    unsigned short* wpk2 = wpk1 + 8 * 16 * 32 * 16;           // 5*2*32*16 bf16

    const int T = 256;
    const int gemmBlocks = (int)(N / 16);         // N % 16 == 0
    const int edgeBlocks = (E + T - 1) / T;
    const long long statWords = N * 80;           // menc + dsum + acc contiguous
    const int statBlocks = (int)((statWords + T - 1) / T);
    const size_t sh1 = (size_t)16 * (512 + 8) * sizeof(unsigned short);
    const size_t sh2 = (size_t)16 * (64 + 8) * sizeof(unsigned short);

    // ----- one-shot weight packing (bf16, WMMA B lane layout) -----
    gat_pack_w<<<dim3(8, 16), 32, 0, stream>>>(W1s, W1d, wpk1, 512, 64, 64);
    gat_pack_w<<<dim3(5, 2),  32, 0, stream>>>(W2s, W2d, wpk2,  64, 40, 40);

    // ----- layer 1 -----
    gat_wmma_proj<<<gemmBlocks, T, sh1, stream>>>(x, wpk1, b1s, b1d, Y, 512, 64, 64);
    gat_zero<<<statBlocks, T, 0, stream>>>((float*)menc, statWords);
    gat_edge_score1<<<edgeBlocks, T, 0, stream>>>(Y, src, dst, a1, score, menc, E);
    gat_edge_acc1<<<edgeBlocks, T, 0, stream>>>(Y, src, dst, score, menc, dsum, acc, E);
    {
        long long n = N * 64;
        gat_final1<<<(int)((n + T - 1) / T), T, 0, stream>>>(acc, dsum, n);
    }

    // ----- layer 2 (hidden h lives in `acc`) -----
    gat_wmma_proj<<<gemmBlocks, T, sh2, stream>>>(acc, wpk2, b2s, b2d, Y, 64, 40, 40);
    gat_zero<<<statBlocks, T, 0, stream>>>((float*)menc, statWords);
    gat_edge_score2<<<edgeBlocks, T, 0, stream>>>(Y, src, dst, a2, score, menc, E);
    gat_edge_acc2<<<edgeBlocks, T, 0, stream>>>(Y, src, dst, score, menc, dsum, acc, E);
    {
        long long n = N * 40;
        gat_final2<<<(int)((n + T - 1) / T), T, 0, stream>>>(acc, dsum, out, n);
    }
}
